// CNFLayer2_24507083391230
// MI455X (gfx1250) — matrix operand order repaired and checked by reading, not compiled, not run on hardware
//
#include <hip/hip_runtime.h>
#include <hip/hip_bf16.h>

// Problem constants (fixed by the reference).
#define N_LIT    200000
#define N_CLAUSE 400000
#define N_EDGE   1600000
#define DIM      128
#define HLS      136   // h_lit row stride in floats (129 used, padded to 32B alignment)

typedef __attribute__((ext_vector_type(16))) __bf16 v16bf;
typedef __attribute__((ext_vector_type(8)))  float  v8f;

// ---------------------------------------------------------------------------
// Zero a float4 region (grid-stride).
// ---------------------------------------------------------------------------
__global__ __launch_bounds__(256) void zero_f4(float4* __restrict__ p, size_t n4) {
    size_t i = (size_t)blockIdx.x * blockDim.x + threadIdx.x;
    size_t stride = (size_t)gridDim.x * blockDim.x;
    float4 z; z.x = 0.f; z.y = 0.f; z.z = 0.f; z.w = 0.f;
    for (; i < n4; i += stride) p[i] = z;
}

// ---------------------------------------------------------------------------
// Pack a 128x128 fp32 weight (row-major W[k][n]) into the B-fragment layout of
// v_wmma_f32_16x16x32_bf16, split into bf16 hi + lo parts.
// Fragment element f of lane L for (ntile t, kstep s):
//   n = t*16 + (L%16),  k = s*32 + (f<8 ? 8*(L/16) + f : 16 + 8*(L/16) + (f-8))
// Stored densely: idx = (((t*4+s)*32 + L)*16 + f)
// ---------------------------------------------------------------------------
__global__ __launch_bounds__(256) void pack_w(const float* __restrict__ W,
                                              __bf16* __restrict__ hi,
                                              __bf16* __restrict__ lo) {
    int idx = blockIdx.x * blockDim.x + threadIdx.x;   // 16384 threads total
    if (idx >= 8 * 4 * 32 * 16) return;
    int f = idx & 15;
    int L = (idx >> 4) & 31;
    int s = (idx >> 9) & 3;
    int t = idx >> 11;
    int n = t * 16 + (L & 15);
    int k = s * 32 + ((f < 8) ? 8 * (L >> 4) + f : 16 + 8 * (L >> 4) + (f - 8));
    float w = W[(size_t)k * DIM + n];
    __bf16 whi = (__bf16)w;
    hi[idx] = whi;
    lo[idx] = (__bf16)(w - (float)whi);
}

// ---------------------------------------------------------------------------
// Scatter 1: h_clause[ec[e]][0:128] += lit_feat[el[e]][0:128]
// One wave per edge; each lane moves 4 consecutive floats (512B coalesced).
// ---------------------------------------------------------------------------
__global__ __launch_bounds__(256) void scatter1(const float* __restrict__ lit_feat,
                                                const int* __restrict__ el,
                                                const int* __restrict__ ec,
                                                float* __restrict__ h_clause) {
    int lane = threadIdx.x & 31;
    int e = __builtin_amdgcn_readfirstlane((int)(blockIdx.x * 8 + (threadIdx.x >> 5)));
    int l = el[e];
    int c = ec[e];
    const float4 v = *(const float4*)(lit_feat + (size_t)l * DIM + lane * 4);
    float* dst = h_clause + (size_t)c * DIM + lane * 4;
    unsafeAtomicAdd(dst + 0, v.x);
    unsafeAtomicAdd(dst + 1, v.y);
    unsafeAtomicAdd(dst + 2, v.z);
    unsafeAtomicAdd(dst + 3, v.w);
}

// ---------------------------------------------------------------------------
// Scatter 2: h_lit[el[e]][0:128] += cembs[ec[e]][0:128];
//            h_lit[el[e]][128]   += clause_feat[ec[e]]
// ---------------------------------------------------------------------------
__global__ __launch_bounds__(256) void scatter2(const float* __restrict__ cembs,
                                                const float* __restrict__ clause_feat,
                                                const int* __restrict__ el,
                                                const int* __restrict__ ec,
                                                float* __restrict__ h_lit) {
    int lane = threadIdx.x & 31;
    int e = __builtin_amdgcn_readfirstlane((int)(blockIdx.x * 8 + (threadIdx.x >> 5)));
    int l = el[e];
    int c = ec[e];
    const float4 v = *(const float4*)(cembs + (size_t)c * DIM + lane * 4);
    float* dst = h_lit + (size_t)l * HLS + lane * 4;
    unsafeAtomicAdd(dst + 0, v.x);
    unsafeAtomicAdd(dst + 1, v.y);
    unsafeAtomicAdd(dst + 2, v.z);
    unsafeAtomicAdd(dst + 3, v.w);
    if (lane == 0) unsafeAtomicAdd(h_lit + (size_t)l * HLS + DIM, clause_feat[c]);
}

// ---------------------------------------------------------------------------
// Out[M,128] = relu(A[:, :128] @ Wpacked + (Wlast ? A[:,128] x Wlast : 0) + bias)
// Split-bf16 WMMA: A = Ahi + Alo (bf16), D += Ahi*Bhi + Alo*Bhi + Ahi*Blo, fp32 acc.
// blockDim = 128 (4 waves); each wave computes one 16-row stripe across all 128 cols.
// No divergence -> EXEC all ones (WMMA requirement).
// ---------------------------------------------------------------------------
__global__ __launch_bounds__(128) void gemm_wmma_relu(const float* __restrict__ A, int lda,
                                                      const __bf16* __restrict__ Bhi,
                                                      const __bf16* __restrict__ Blo,
                                                      const float* __restrict__ Wlast, // [128] or null
                                                      const float* __restrict__ bias,  // [128]
                                                      float* __restrict__ Out) {
    int lane = threadIdx.x & 31;
    int wave = threadIdx.x >> 5;
    int tile = blockIdx.x * 4 + wave;        // 16-row output tile index
    int mlo  = lane & 15;
    int h    = lane >> 4;

    // --- Load & split A fragments for 4 K-steps (ISA 16-bit A layout:
    // lane holds row M=lane%16; a[0..7]=K k0+8h+i, a[8..15]=K k0+16+8h+i) ---
    const float* arow = A + (size_t)(tile * 16 + mlo) * lda;
    v16bf ahi[4], alo[4];
#pragma unroll
    for (int s = 0; s < 4; ++s) {
        int k0 = s * 32 + 8 * h;
#pragma unroll
        for (int i = 0; i < 8; ++i) {
            float x0 = arow[k0 + i];
            float x1 = arow[k0 + 16 + i];
            __bf16 h0 = (__bf16)x0;
            __bf16 h1 = (__bf16)x1;
            ahi[s][i]     = h0;  alo[s][i]     = (__bf16)(x0 - (float)h0);
            ahi[s][8 + i] = h1;  alo[s][8 + i] = (__bf16)(x1 - (float)h1);
        }
    }

    // K=129 extra column (exact fp32 rank-1 term), rows this lane will write.
    float x128[8];
    if (Wlast) {
#pragma unroll
        for (int r = 0; r < 8; ++r)
            x128[r] = A[(size_t)(tile * 16 + 8 * h + r) * lda + DIM];
    }

#pragma unroll
    for (int t = 0; t < 8; ++t) {           // 8 column tiles of 16
        v8f c = {0.f, 0.f, 0.f, 0.f, 0.f, 0.f, 0.f, 0.f};
#pragma unroll
        for (int s = 0; s < 4; ++s) {       // K = 4 x 32
            const v16bf bh = *(const v16bf*)(Bhi + (((t * 4 + s) * 32 + lane) << 4));
            const v16bf bl = *(const v16bf*)(Blo + (((t * 4 + s) * 32 + lane) << 4));
            c = __builtin_amdgcn_wmma_f32_16x16x32_bf16(false, ahi[s], false, bh, (short)0, c, false, false);
            c = __builtin_amdgcn_wmma_f32_16x16x32_bf16(false, alo[s], false, bh, (short)0, c, false, false);
            c = __builtin_amdgcn_wmma_f32_16x16x32_bf16(false, ahi[s], false, bl, (short)0, c, false, false);
        }
        // Epilogue: C/D layout — lane holds N=lane%16, VGPR r holds M = r + 8h.
        int n = t * 16 + mlo;
        float bb = bias[n];
        float wl = Wlast ? Wlast[n] : 0.f;
#pragma unroll
        for (int r = 0; r < 8; ++r) {
            float v = c[r] + bb;
            if (Wlast) v += x128[r] * wl;
            v = v > 0.f ? v : 0.f;
            Out[(size_t)(tile * 16 + 8 * h + r) * DIM + n] = v;
        }
    }
}

// ---------------------------------------------------------------------------
extern "C" void kernel_launch(void* const* d_in, const int* in_sizes, int n_in,
                              void* d_out, int out_size, void* d_ws, size_t ws_size,
                              hipStream_t stream) {
    const float* lit_feat    = (const float*)d_in[0];  // [200000,128]
    const float* clause_feat = (const float*)d_in[1];  // [400000,1]
    const float* W_l2c       = (const float*)d_in[2];  // [128,128]
    const float* b_l2c       = (const float*)d_in[3];  // [128]
    const float* W_c2l       = (const float*)d_in[4];  // [129,128]
    const float* b_c2l       = (const float*)d_in[5];  // [128]
    const int*   edge_lit    = (const int*)d_in[6];    // [1.6M] (harness bridges ints to i32)
    const int*   edge_clause = (const int*)d_in[7];    // [1.6M]
    float*       out         = (float*)d_out;          // [200000,128]

    // Workspace carve-up (all offsets multiples of 256B).
    float*  h_clause = (float*)d_ws;                               // 51,200,000 f
    float*  h_lit    = h_clause + (size_t)N_CLAUSE * DIM;          // 27,200,000 f
    float*  cembs    = h_lit + (size_t)N_LIT * HLS;                // 51,200,000 f
    __bf16* w1hi     = (__bf16*)(cembs + (size_t)N_CLAUSE * DIM);  // 16384 each
    __bf16* w1lo     = w1hi + 16384;
    __bf16* w2hi     = w1lo + 16384;
    __bf16* w2lo     = w2hi + 16384;

    // 1. Zero accumulators (h_clause + h_lit are contiguous).
    size_t n4 = ((size_t)N_CLAUSE * DIM + (size_t)N_LIT * HLS) / 4;
    zero_f4<<<8192, 256, 0, stream>>>((float4*)h_clause, n4);

    // 2. Pack weights into WMMA B-fragment layout (bf16 hi/lo).
    pack_w<<<64, 256, 0, stream>>>(W_l2c, w1hi, w1lo);
    pack_w<<<64, 256, 0, stream>>>(W_c2l, w2hi, w2lo);   // rows 0..127 only

    // 3. Edge scatter #1 (one wave per edge).
    scatter1<<<N_EDGE / 8, 256, 0, stream>>>(lit_feat, edge_lit, edge_clause, h_clause);

    // 4. cembs = relu(h_clause @ W_l2c + b_l2c) via split-bf16 WMMA.
    gemm_wmma_relu<<<N_CLAUSE / 64, 128, 0, stream>>>(h_clause, DIM, w1hi, w1lo,
                                                      nullptr, b_l2c, cembs);

    // 5. Edge scatter #2 (128-wide message + scalar clause feature).
    scatter2<<<N_EDGE / 8, 256, 0, stream>>>(cembs, clause_feat, edge_lit, edge_clause, h_lit);

    // 6. out = relu(h_lit[:, :128] @ W_c2l[:128] + h_lit[:,128] x W_c2l[128] + b_c2l).
    gemm_wmma_relu<<<N_LIT / 64, 128, 0, stream>>>(h_lit, HLS, w2hi, w2lo,
                                                   W_c2l + (size_t)DIM * DIM, b_c2l, out);
}
